// SolverVAE_32916629356746
// MI455X (gfx1250) — matrix operand order repaired and checked by reading, compiled, not run on hardware
//
#include <hip/hip_runtime.h>
#include <hip/hip_bf16.h>
#include <stdint.h>

// ---------------------------------------------------------------------------
// SolverVAE for MI455X (gfx1250): bf16 WMMA encoder GEMMs (TDM-staged A tile,
// 16x64 C tile per wave, scalarized tile count) + wave32 IPM solver
// ---------------------------------------------------------------------------

typedef __attribute__((ext_vector_type(16))) __bf16 v16bf;
typedef __attribute__((ext_vector_type(8)))  __bf16 v8bf;
typedef __attribute__((ext_vector_type(8)))  float  v8f;
typedef __attribute__((ext_vector_type(4)))  unsigned int v4u;
typedef __attribute__((ext_vector_type(4)))  int v4i;
typedef __attribute__((ext_vector_type(8)))  int v8i;

constexpr int kB    = 1024;   // batch
constexpr int kM    = 32;     // constraints
constexpr int kN    = 128;    // LP variables (y_dim)
constexpr int kYD   = 128;
constexpr int kXD   = 64;
constexpr int kOBS  = kYD + kXD;       // 192
constexpr int kWDIM = (kYD - kM) * kM; // 96*32 = 3072

// ---------------------------- small device utils ---------------------------

__device__ __forceinline__ uint32_t pcg_hash(uint32_t v) {
  v = v * 747796405u + 2891336453u;
  uint32_t w = ((v >> ((v >> 28) + 4u)) ^ v) * 277803737u;
  return (w >> 22) ^ w;
}
__device__ __forceinline__ float u01(uint32_t h) {
  return ((h >> 8) + 0.5f) * (1.0f / 16777216.0f);
}
__device__ __forceinline__ float siluf(float v) { return v / (1.f + __expf(-v)); }
__device__ __forceinline__ float softplusf(float v) {
  return (v > 20.f) ? v : log1pf(__expf(v));
}
__device__ __forceinline__ float wave_sum32(float v) {
  for (int o = 16; o > 0; o >>= 1) v += __shfl_xor(v, o, 32);
  return v;
}
__device__ __forceinline__ float wave_min32(float v) {
  for (int o = 16; o > 0; o >>= 1) v = fminf(v, __shfl_xor(v, o, 32));
  return v;
}

// --------------------------- data movement kernels -------------------------

__global__ void cvt_f32_bf16_kernel(const float* __restrict__ in,
                                    __bf16* __restrict__ out, int count) {
  int i = blockIdx.x * blockDim.x + threadIdx.x;
  if (i < count) out[i] = (__bf16)in[i];
}

// obs = concat([y, x], axis=1): [B, 192] in bf16
__global__ void concat_obs_kernel(const float* __restrict__ y,
                                  const float* __restrict__ x,
                                  __bf16* __restrict__ out) {
  int i = blockIdx.x * blockDim.x + threadIdx.x;
  if (i >= kB * kOBS) return;
  int b = i / kOBS, j = i % kOBS;
  float v = (j < kYD) ? y[b * kYD + j] : x[b * kXD + (j - kYD)];
  out[i] = (__bf16)v;
}

// Pack fp32 weight W[K][N] (row-major) into WMMA B-fragment-major bf16:
// element p = ((kt*(N/16)+nt)*32 + lane)*16 + h holds
//   W[kt*32 + (h>=8 ? 16:0) + (lane>=16 ? 8:0) + (h&7)][nt*16 + (lane&15)]
// so each lane of a wave loads its whole v16bf fragment as one contiguous 32B.
__global__ void pack_w_kernel(const float* __restrict__ w,
                              __bf16* __restrict__ out, int K, int N) {
  int p = blockIdx.x * blockDim.x + threadIdx.x;
  if (p >= K * N) return;
  int h    = p & 15;
  int lane = (p >> 4) & 31;
  int tile = p >> 9;
  int ntN  = N >> 4;
  int nt   = tile % ntN;
  int kt   = tile / ntN;
  int k    = kt * 32 + ((h & 8) ? 16 : 0) + ((lane & 16) ? 8 : 0) + (h & 7);
  int c    = nt * 16 + (lane & 15);
  out[p] = (__bf16)w[(size_t)k * N + c];
}

// ------------------------------- WMMA GEMM ---------------------------------
// Straight-line per-wave worker: NV accumulators (16x16 each) sharing one
// LDS A fragment per 32-K step. No divergent control flow around WMMA.
template <int NV>
__device__ __forceinline__ void gemm_tile(
    const __bf16* __restrict__ As, const __bf16* __restrict__ Wp,
    const float* __restrict__ bias, __bf16* __restrict__ Ybf,
    float* __restrict__ Yf, int K, int N, int act, int ntN, int nt0, int mt,
    int lane) {
  const int rowl = lane & 15;            // A: M = lane%16 in both halves
  const int b0   = (lane < 16) ? 0 : 8;  // A half-wave K sub-offset
  const __bf16* arow = As + (size_t)rowl * K;
  v8f acc[NV] = {};
  for (int k0 = 0; k0 < K; k0 += 32) {
    v8bf a_lo = *(const v8bf*)(arow + k0 + b0);
    v8bf a_hi = *(const v8bf*)(arow + k0 + 16 + b0);
    v16bf af = __builtin_shufflevector(a_lo, a_hi,
        0, 1, 2, 3, 4, 5, 6, 7, 8, 9, 10, 11, 12, 13, 14, 15);
    const __bf16* bbase =
        Wp + (((size_t)(k0 >> 5) * ntN + nt0) * 32 + lane) * 16;
    if (k0 + 64 < K)  // stream hint for the B weights two K-tiles ahead
      __builtin_prefetch(Wp + ((size_t)((k0 >> 5) + 2) * ntN + nt0) * 512, 0, 1);
#pragma unroll
    for (int u = 0; u < NV; ++u) {
      v16bf bf_ = *(const v16bf*)(bbase + (size_t)u * 512);
      acc[u] = __builtin_amdgcn_wmma_f32_16x16x32_bf16(
          false, af, false, bf_, (short)0, acc[u], false, false);
    }
  }
  // epilogue: C layout VGPR i -> M = i + (lane>=16 ? 8:0), N = lane%16
  const int colb  = lane & 15;
  const int rbase = mt * 16 + ((lane < 16) ? 0 : 8);
#pragma unroll
  for (int u = 0; u < NV; ++u) {
    const int col  = (nt0 + u) * 16 + colb;
    const float bv = bias ? bias[col] : 0.f;
#pragma unroll
    for (int i = 0; i < 8; i++) {
      float v = acc[u][i] + bv;
      if (act == 1)      v = siluf(v);
      else if (act == 2) v = softplusf(v);
      size_t idx = (size_t)(rbase + i) * N + col;
      Ybf[idx] = (__bf16)v;
      if (Yf) Yf[idx] = v;
    }
  }
}

// Y[M,N] = act(A[M,K] @ W[K,N] + bias)  ; A bf16 row-major, W packed bf16.
// Block: 128 threads = 4 waves. The block's 16-row x K A-strip (contiguous in
// memory) is staged to LDS once via the Tensor Data Mover; each wave then owns
// up to a 16x64 C tile. act: 0=none 1=silu 2=softplus.
__global__ __launch_bounds__(128)
void gemm_bf16_kernel(const __bf16* __restrict__ A, const __bf16* __restrict__ Wp,
                      const float* __restrict__ bias, __bf16* __restrict__ Ybf,
                      float* __restrict__ Yf, int M, int K, int N, int act) {
  extern __shared__ __bf16 As[];  // 16 x K strip, row-major (contiguous copy)
  const int lane = threadIdx.x & 31;
  const int ntN  = N >> 4;
  const int mt   = blockIdx.y;

  // ---- stage A strip: rows [mt*16, mt*16+16) x K, contiguous 16*K elements
#if defined(__has_builtin) && __has_builtin(__builtin_amdgcn_tensor_load_to_lds)
  if (threadIdx.x < 32) {  // wave 0 issues one TDM descriptor (EXEC-agnostic)
    uint64_t ga = (uint64_t)(uintptr_t)(A + (size_t)mt * 16 * K);
    uint32_t ldsOff = (uint32_t)(uintptr_t)(void*)As;  // low 32b = LDS offset
    uint32_t total = (uint32_t)(16 * K);               // elements (< 65536)
    // D# group0: count=1 | lds_addr | global_addr[56:0] | type=2
    v4u g0 = {1u, ldsOff, (unsigned)ga,
              (unsigned)(((ga >> 32) & 0x1FFFFFFu) | (2u << 30))};
    // D# group1: data_size=1(2B); tensor_dim0=total; tensor_dim1=1;
    //            tile_dim0=total; tile_dim1=1; stride0=total
    v8i g1 = {(int)(1u << 16),
              (int)((total & 0xFFFFu) << 16),
              (int)(((total >> 16) & 0xFFFFu) | (1u << 16)),
              (int)(total << 16),
              (int)1,
              (int)total,
              0, 0};
    v4i z4 = {0, 0, 0, 0};
#if __clang_major__ >= 23
    v8i z8 = {0, 0, 0, 0, 0, 0, 0, 0};
    __builtin_amdgcn_tensor_load_to_lds(g0, g1, z4, z4, z8, 0);
#else
    __builtin_amdgcn_tensor_load_to_lds(g0, g1, z4, z4, 0);
#endif
    __builtin_amdgcn_s_wait_tensorcnt(0);
  }
#else
  {  // fallback: cooperative vector copy
    const int elems = 16 * K;
    for (int i = threadIdx.x * 8; i < elems; i += 128 * 8)
      *(v8bf*)(As + i) = *(const v8bf*)(A + (size_t)mt * 16 * K + i);
  }
#endif
  __syncthreads();

  // ---- per-wave subtile count, forced into an SGPR so the dispatch below is
  // scalar control flow (keeps EXEC all-ones through every WMMA).
  const int nt0 = (blockIdx.x * 4 + (threadIdx.x >> 5)) * 4;
  int nvalid = ntN - nt0;
  if (nvalid < 0) nvalid = 0;
  if (nvalid > 4) nvalid = 4;
  nvalid = __builtin_amdgcn_readfirstlane(nvalid);

  switch (nvalid) {
    case 4: gemm_tile<4>(As, Wp, bias, Ybf, Yf, K, N, act, ntN, nt0, mt, lane); break;
    case 3: gemm_tile<3>(As, Wp, bias, Ybf, Yf, K, N, act, ntN, nt0, mt, lane); break;
    case 2: gemm_tile<2>(As, Wp, bias, Ybf, Yf, K, N, act, ntN, nt0, mt, lane); break;
    case 1: gemm_tile<1>(As, Wp, bias, Ybf, Yf, K, N, act, ntN, nt0, mt, lane); break;
    default: break;  // wave has no output columns
  }
}

// ------------------------------- sampling ----------------------------------

// W_sample[b, i, j] : j<96 -> sigmoid(Wl[b,i,j] + binary-concrete noise)
//                     j>=96 -> identity I[i, j-96]
__global__ void sample_W_kernel(const float* __restrict__ Wl,
                                float* __restrict__ Wout) {
  int idx = blockIdx.x * blockDim.x + threadIdx.x;
  if (idx >= kB * kM * kN) return;
  int j = idx & (kN - 1);
  int i = (idx >> 7) & (kM - 1);
  int b = idx >> 12;
  float v;
  if (j < kN - kM) {
    float logit = Wl[(size_t)b * kWDIM + i * (kN - kM) + j];
    float u = u01(pcg_hash(0x1234567u ^ (uint32_t)idx));
    u = fminf(fmaxf(u, 1e-6f), 1.f - 1e-6f);
    v = 1.f / (1.f + __expf(-(logit + __logf(u) - log1pf(-u))));
  } else {
    v = (i == (j - (kN - kM))) ? 1.f : 0.f;
  }
  Wout[idx] = v;
}

// Gamma(shape)/rate via Marsaglia-Tsang (bounded attempts, deterministic RNG)
__global__ void gamma_sample_kernel(const float* __restrict__ shape,
                                    const float* __restrict__ rate,
                                    float* __restrict__ out, int count,
                                    float clampMin, unsigned seed) {
  int idx = blockIdx.x * blockDim.x + threadIdx.x;
  if (idx >= count) return;
  float a = fmaxf(shape[idx], 1e-8f);
  float r = fmaxf(rate[idx], 1e-8f);
  unsigned s0 = seed ^ (unsigned)((uint32_t)idx * 2654435761u);
  float boost = 1.f;
  if (a < 1.f) {
    float u = u01(pcg_hash(s0 ^ 0x9e3779b9u));
    boost = powf(u, 1.f / a);
    a += 1.f;
  }
  float d = a - (1.f / 3.f);
  float c = rsqrtf(9.f * d);
  float g = d;  // fallback = mode/mean-ish
  for (int i = 0; i < 8; i++) {
    unsigned h1 = pcg_hash(s0 + (unsigned)(i * 0x85ebca6bu) + 1u);
    unsigned h2 = pcg_hash(h1 ^ 0xc2b2ae35u);
    unsigned h3 = pcg_hash(h2 + 0x27d4eb2fu);
    float z = sqrtf(-2.f * __logf(u01(h1))) * __cosf(6.28318530718f * u01(h2));
    float v = 1.f + c * z;
    if (v <= 0.f) continue;
    v = v * v * v;
    if (__logf(u01(h3)) < 0.5f * z * z + d - d * v + d * __logf(v)) {
      g = d * v;
      break;
    }
  }
  float val = g * boost / r;
  if (clampMin > 0.f) val = fmaxf(val, clampMin);
  out[idx] = val;
}

// ------------------------------ IPM LP solver ------------------------------
// One wave32 block per batch sample. m=32 (== wave size), n=128 (4 per lane).
// 25 iterations of infeasible primal-dual IPM with normal-equations Cholesky.
__global__ __launch_bounds__(32)
void ipm_kernel(const float* __restrict__ Wg, const float* __restrict__ hg,
                const float* __restrict__ qg, float* __restrict__ yout,
                float* __restrict__ Qout) {
  __shared__ float Ws[kM * kN];        // W row-major
  __shared__ float Ms[kM * 33];        // M / Cholesky L (padded rows)
  __shared__ float xv[kN], sv[kN], dvv[kN], rdv[kN], gvv[kN], rcv[kN];
  __shared__ float dsv[kN], dxv[kN], qv[kN];
  __shared__ float lamv[kM], rhsv[kM], zv[kM], hv[kM];
  const int t = threadIdx.x;
  const int b = blockIdx.x;

  for (int nn = t; nn < kN; nn += 32) {
    qv[nn] = qg[(size_t)b * kN + nn];
    xv[nn] = 1.f;
    sv[nn] = 1.f;
  }
  for (int nn = 0; nn < kN; nn++)
    Ws[t * kN + nn] = Wg[(size_t)b * (kM * kN) + t * kN + nn];
  hv[t] = hg[(size_t)b * kM + t];
  lamv[t] = 0.f;
  __syncthreads();

  for (int it = 0; it < 25; it++) {
    float ps = 0.f;
    for (int k = 0; k < 4; k++) { int nn = t * 4 + k; ps += xv[nn] * sv[nn]; }
    float mu = wave_sum32(ps) * (1.f / (float)kN);
    float rp = -hv[t];
    for (int nn = 0; nn < kN; nn++) rp += Ws[t * kN + nn] * xv[nn];
    for (int k = 0; k < 4; k++) {
      int nn = t * 4 + k;
      float a = sv[nn] - qv[nn];
      for (int mm = 0; mm < kM; mm++) a += Ws[mm * kN + nn] * lamv[mm];
      rdv[nn] = a;
      float rc = xv[nn] * sv[nn] - 0.1f * mu;
      rcv[nn] = rc;
      dvv[nn] = xv[nn] / sv[nn];
      gvv[nn] = (xv[nn] * a - rc) / sv[nn];
    }
    __syncthreads();
    for (int kk = 0; kk < kM; kk++) {
      float a = 0.f;
      for (int nn = 0; nn < kN; nn++)
        a += Ws[t * kN + nn] * dvv[nn] * Ws[kk * kN + nn];
      Ms[t * 33 + kk] = a + ((t == kk) ? 1e-6f : 0.f);
    }
    float rh = -rp;
    for (int nn = 0; nn < kN; nn++) rh -= Ws[t * kN + nn] * gvv[nn];
    rhsv[t] = rh;
    __syncthreads();
    for (int j = 0; j < kM; j++) {
      if (t == j) {
        float a = Ms[j * 33 + j];
        for (int k = 0; k < j; k++) a -= Ms[j * 33 + k] * Ms[j * 33 + k];
        Ms[j * 33 + j] = sqrtf(fmaxf(a, 1e-30f));
      }
      __syncthreads();
      if (t > j) {
        float a = Ms[t * 33 + j];
        for (int k = 0; k < j; k++) a -= Ms[t * 33 + k] * Ms[j * 33 + k];
        Ms[t * 33 + j] = a / Ms[j * 33 + j];
      }
      __syncthreads();
    }
    zv[t] = rhsv[t];
    __syncthreads();
    for (int j = 0; j < kM; j++) {
      if (t == j) zv[j] /= Ms[j * 33 + j];
      __syncthreads();
      if (t > j) zv[t] -= Ms[t * 33 + j] * zv[j];
      __syncthreads();
    }
    for (int j = kM - 1; j >= 0; j--) {
      if (t == j) zv[j] /= Ms[j * 33 + j];
      __syncthreads();
      if (t < j) zv[t] -= Ms[j * 33 + t] * zv[j];
      __syncthreads();
    }
    float amin = 1e20f;
    for (int k = 0; k < 4; k++) {
      int nn = t * 4 + k;
      float dsx = -rdv[nn];
      for (int mm = 0; mm < kM; mm++) dsx -= Ws[mm * kN + nn] * zv[mm];
      float dxx = (-rcv[nn] - xv[nn] * dsx) / sv[nn];
      dsv[nn] = dsx;
      dxv[nn] = dxx;
      if (dxx < 0.f) amin = fminf(amin, -xv[nn] / dxx);
      if (dsx < 0.f) amin = fminf(amin, -sv[nn] / dsx);
    }
    float alpha = 0.99f * fminf(1.f, wave_min32(amin));
    __syncthreads();
    for (int k = 0; k < 4; k++) {
      int nn = t * 4 + k;
      xv[nn] += alpha * dxv[nn];
      sv[nn] += alpha * dsv[nn];
    }
    lamv[t] += alpha * zv[t];
    __syncthreads();
  }

  float ps = 0.f;
  for (int k = 0; k < 4; k++) {
    int nn = t * 4 + k;
    yout[(size_t)b * kN + nn] = xv[nn];
    ps += qv[nn] * xv[nn];
  }
  float Q = wave_sum32(ps);
  if (t == 0) Qout[b] = Q;
}

// ------------------------------- host driver -------------------------------

// d_in flatten order (jax pytree: dict keys sorted, {w,b} -> [b,w]):
// 0:y 1:x, then prior_params:
//   2:Wl1.b 3:Wl1.w 4:Wl2.b 5:Wl2.w 6:h1.b 7:h1.w 8:h_rate.b 9:h_rate.w
//   10:h_shape.b 11:h_shape.w 12..19: mlp[0..3].{b,w} 20:q1.b 21:q1.w
//   22:q_rate.b 23:q_rate.w 24:q_shape.b 25:q_shape.w
// then recog_params (3 mlp layers): 26..47 analogous.
extern "C" void kernel_launch(void* const* d_in, const int* in_sizes, int n_in,
                              void* d_out, int out_size, void* d_ws,
                              size_t ws_size, hipStream_t stream) {
  (void)in_sizes; (void)n_in; (void)out_size; (void)ws_size;
  const float* y = (const float*)d_in[0];
  const float* x = (const float*)d_in[1];
  auto F = [&](int i) { return (const float*)d_in[i]; };

  char* ws = (char*)d_ws;
  size_t off = 0;
  auto take = [&](size_t bytes) -> void* {
    off = (off + 255) & ~(size_t)255;
    void* p = ws + off;
    off += bytes;
    return p;
  };

  auto packW = [&](const float* w, int K, int N) -> __bf16* {
    __bf16* dst = (__bf16*)take((size_t)K * N * sizeof(__bf16));
    int total = K * N;
    pack_w_kernel<<<(total + 255) / 256, 256, 0, stream>>>(w, dst, K, N);
    return dst;
  };
  auto gemm = [&](const __bf16* A, const __bf16* Wp, const float* bias,
                  __bf16* Ybf, float* Yf, int M, int K, int N, int act) {
    dim3 g((N + 255) / 256, M / 16);
    size_t lds = (size_t)16 * K * sizeof(__bf16);  // A strip
    gemm_bf16_kernel<<<g, 128, lds, stream>>>(A, Wp, bias, Ybf, Yf, M, K, N, act);
  };

  // packed weights (converted every launch; ~6us of HBM traffic at 23.3TB/s)
  __bf16* p_m0 = packW(F(13), kXD, 128);
  __bf16* p_m1 = packW(F(15), 128, 256);
  __bf16* p_m2 = packW(F(17), 256, 512);
  __bf16* p_m3 = packW(F(19), 512, 1024);
  __bf16* p_Wl1 = packW(F(3), 1024, kWDIM);
  __bf16* p_Wl2 = packW(F(5), kWDIM, kWDIM);
  __bf16* p_h1 = packW(F(7), 1024, 1024);
  __bf16* p_hsh = packW(F(11), 1024, kM);
  __bf16* p_hrt = packW(F(9), 1024, kM);
  __bf16* p_q1 = packW(F(21), 1024, 1024);
  __bf16* p_qsh = packW(F(25), 1024, kYD);
  __bf16* p_qrt = packW(F(23), 1024, kYD);

  __bf16* r_m0 = packW(F(37), kOBS, 256);
  __bf16* r_m1 = packW(F(39), 256, 512);
  __bf16* r_m2 = packW(F(41), 512, 1024);
  __bf16* r_Wl1 = packW(F(27), 1024, kWDIM);
  __bf16* r_Wl2 = packW(F(29), kWDIM, kWDIM);
  __bf16* r_h1 = packW(F(31), 1024, 1024);
  __bf16* r_hsh = packW(F(35), 1024, kM);
  __bf16* r_hrt = packW(F(33), 1024, kM);
  __bf16* r_q1 = packW(F(43), 1024, 1024);
  __bf16* r_qsh = packW(F(47), 1024, kYD);
  __bf16* r_qrt = packW(F(45), 1024, kYD);

  // activation buffers (bf16)
  __bf16* xb     = (__bf16*)take((size_t)kB * kXD * 2);
  __bf16* obsb   = (__bf16*)take((size_t)kB * kOBS * 2);
  __bf16* t0     = (__bf16*)take((size_t)kB * 512 * 2);
  __bf16* t1     = (__bf16*)take((size_t)kB * 512 * 2);
  __bf16* hB     = (__bf16*)take((size_t)kB * 1024 * 2);
  __bf16* hhB    = (__bf16*)take((size_t)kB * 1024 * 2);
  __bf16* big0   = (__bf16*)take((size_t)kB * kWDIM * 2);
  __bf16* big1   = (__bf16*)take((size_t)kB * kWDIM * 2);
  __bf16* headbf = (__bf16*)take((size_t)kB * kYD * 2);

  // output segments (reference tuple order, flat fp32)
  float* out = (float*)d_out;
  const size_t S_Wl = (size_t)kB * kWDIM, S_h = (size_t)kB * kM,
               S_q = (size_t)kB * kYD;
  float* o_pWl = out;
  float* o_phs = o_pWl + S_Wl;
  float* o_phr = o_phs + S_h;
  float* o_pqs = o_phr + S_h;
  float* o_pqr = o_pqs + S_q;
  float* o_rWl = o_pqr + S_q;
  float* o_rhs_ = o_rWl + S_Wl;
  float* o_rhr = o_rhs_ + S_h;
  float* o_rqs = o_rhr + S_h;
  float* o_rqr = o_rqs + S_q;
  float* o_ys = o_rqr + S_q;               // y_sample  [B,128]
  float* o_Wsamp = o_ys + (size_t)kB * kN; // W_sample  [B,32,128]
  float* o_hsamp = o_Wsamp + (size_t)kB * kM * kN; // h_sample [B,32]
  float* o_qsamp = o_hsamp + S_h;          // q_sample  [B,128]
  float* o_Q = o_qsamp + S_q;              // Q_sample  [B]

  // --- inputs to bf16 ---
  cvt_f32_bf16_kernel<<<(kB * kXD + 255) / 256, 256, 0, stream>>>(x, xb, kB * kXD);
  concat_obs_kernel<<<(kB * kOBS + 255) / 256, 256, 0, stream>>>(y, x, obsb);

  // --- prior encoder ---
  gemm(xb, p_m0, F(12), t0, nullptr, kB, kXD, 128, 1);
  gemm(t0, p_m1, F(14), t1, nullptr, kB, 128, 256, 1);
  gemm(t1, p_m2, F(16), t0, nullptr, kB, 256, 512, 1);
  gemm(t0, p_m3, F(18), hB, nullptr, kB, 512, 1024, 1);
  gemm(hB, p_Wl1, F(2), big0, nullptr, kB, 1024, kWDIM, 1);
  gemm(big0, p_Wl2, F(4), big1, o_pWl, kB, kWDIM, kWDIM, 1);
  gemm(hB, p_h1, F(6), hhB, nullptr, kB, 1024, 1024, 1);
  gemm(hhB, p_hsh, F(10), headbf, o_phs, kB, 1024, kM, 2);
  gemm(hhB, p_hrt, F(8), headbf, o_phr, kB, 1024, kM, 2);
  gemm(hB, p_q1, F(20), hhB, nullptr, kB, 1024, 1024, 1);
  gemm(hhB, p_qsh, F(24), headbf, o_pqs, kB, 1024, kYD, 2);
  gemm(hhB, p_qrt, F(22), headbf, o_pqr, kB, 1024, kYD, 2);

  // --- recognition encoder (posterior) ---
  gemm(obsb, r_m0, F(36), t0, nullptr, kB, kOBS, 256, 1);
  gemm(t0, r_m1, F(38), t1, nullptr, kB, 256, 512, 1);
  gemm(t1, r_m2, F(40), hB, nullptr, kB, 512, 1024, 1);
  gemm(hB, r_Wl1, F(26), big0, nullptr, kB, 1024, kWDIM, 1);
  gemm(big0, r_Wl2, F(28), big1, o_rWl, kB, kWDIM, kWDIM, 1);
  gemm(hB, r_h1, F(30), hhB, nullptr, kB, 1024, 1024, 1);
  gemm(hhB, r_hsh, F(34), headbf, o_rhs_, kB, 1024, kM, 2);
  gemm(hhB, r_hrt, F(32), headbf, o_rhr, kB, 1024, kM, 2);
  gemm(hB, r_q1, F(42), hhB, nullptr, kB, 1024, 1024, 1);
  gemm(hhB, r_qsh, F(46), headbf, o_rqs, kB, 1024, kYD, 2);
  gemm(hhB, r_qrt, F(44), headbf, o_rqr, kB, 1024, kYD, 2);

  // --- sampling ---
  sample_W_kernel<<<(kB * kM * kN + 255) / 256, 256, 0, stream>>>(o_pWl, o_Wsamp);
  gamma_sample_kernel<<<((int)S_h + 255) / 256, 256, 0, stream>>>(
      o_phs, o_phr, o_hsamp, (int)S_h, 0.f, 0xA341316Cu);
  gamma_sample_kernel<<<((int)S_q + 255) / 256, 256, 0, stream>>>(
      o_pqs, o_pqr, o_qsamp, (int)S_q, 0.001f, 0xC8013EA4u);

  // --- interior point LP solve: y_sample, Q_sample ---
  ipm_kernel<<<kB, 32, 0, stream>>>(o_Wsamp, o_hsamp, o_qsamp, o_ys, o_Q);
}
